// MixER_15109694947507
// MI455X (gfx1250) — matrix operand order repaired and checked by reading, not compiled, not run on hardware
//
#include <hip/hip_runtime.h>

// ---------------------------------------------------------------- constants
constexpr int   kDATA  = 64;
constexpr int   kCTX   = 128;
constexpr int   kNEXP  = 8;
constexpr int   kENVS  = 16;
constexpr int   kNTOK  = 2048;
constexpr long  kNET   = 164672;   // per-expert flattened param count

// flattened per-expert offsets (match reference OFFS)
constexpr long OW1 = 0,      OB1 = 16384;
constexpr long OW2 = 16640,  OB2 = 82176;
constexpr long OW3 = 82432,  OB3 = 147968;
constexpr long OW4 = 148224, OB4 = 164608;

// ---------------------------------------------------------------- WMMA types
typedef __attribute__((ext_vector_type(16))) __bf16 v16bf;
typedef __attribute__((ext_vector_type(8)))  float  v8f;
typedef __attribute__((ext_vector_type(2)))  __bf16 v2bf;
typedef __attribute__((ext_vector_type(4)))  int    i32x4;

// ------------------------------------------------- bf16 conversion (native)
__device__ __forceinline__ unsigned short f2bfu(float f) {
  __bf16 h = (__bf16)f;                       // fptrunc, RNE
  return __builtin_bit_cast(unsigned short, h);
}
__device__ __forceinline__ unsigned pack_bf2(float a, float b) {
#if __has_builtin(__builtin_amdgcn_cvt_pk_bf16_f32)
  v2bf p = __builtin_amdgcn_cvt_pk_bf16_f32(a, b);
  return __builtin_bit_cast(unsigned, p);
#else
  return (unsigned)f2bfu(a) | ((unsigned)f2bfu(b) << 16);
#endif
}
__device__ __forceinline__ float bfu2f(unsigned short s) {
  return __uint_as_float(((unsigned)s) << 16);
}

// ------------------------------------------------- async global->LDS staging
#if __has_builtin(__builtin_amdgcn_global_load_async_to_lds_b128) && \
    __has_builtin(__builtin_amdgcn_s_wait_asynccnt)
#define USE_ASYNC_LDS 1
#else
#define USE_ASYNC_LDS 0
#endif

__device__ __forceinline__ void copy16g(const void* g, void* l) {
#if USE_ASYNC_LDS
  __builtin_amdgcn_global_load_async_to_lds_b128(
      (__attribute__((address_space(1))) i32x4*)(void*)g,
      (__attribute__((address_space(3))) i32x4*)l, 0, 0);
#else
  *reinterpret_cast<uint4*>(l) = *reinterpret_cast<const uint4*>((const void*)g);
#endif
}
__device__ __forceinline__ void wait_async_lds() {
#if USE_ASYNC_LDS
  __builtin_amdgcn_s_wait_asynccnt(0);
#endif
}

// ----------------------------------------------------- WMMA fragment loaders
// A fragment: 16x32 bf16 (MxK) from row-major tile, leading dim ld (padded).
__device__ __forceinline__ v16bf load_a_frag(const unsigned short* base, int ld,
                                             int lane, int kbase) {
  v16bf a;
  unsigned* ap = reinterpret_cast<unsigned*>(&a);
  const int m  = lane & 15;
  const int hi = lane >> 4;
  const unsigned short* row = base + m * ld;
#pragma unroll
  for (int ii = 0; ii < 8; ++ii) {
    int i = ii * 2;
    int k = (i & 7) + ((i >> 3) << 4) + (hi << 3) + kbase;
    ap[ii] = *reinterpret_cast<const unsigned*>(row + k);
  }
  return a;
}
// B fragment: 32x16 bf16 (KxN) from row-major W tile [N][ld]: B[k][n]=W[n][k]
__device__ __forceinline__ v16bf load_b_frag(const unsigned short* wtile, int ld,
                                             int lane, int kbase) {
  v16bf b;
  unsigned* bp = reinterpret_cast<unsigned*>(&b);
  const int n  = lane & 15;
  const int hi = lane >> 4;
  const unsigned short* row = wtile + n * ld;
#pragma unroll
  for (int ii = 0; ii < 8; ++ii) {
    int k = ((ii * 2) & 7) + (((ii * 2) >> 3) << 4) + (hi << 3) + kbase;
    bp[ii] = *reinterpret_cast<const unsigned*>(row + k);
  }
  return b;
}

// ---------------------------------------------------------------- tiny prep
__global__ void concat_base_kernel(const float* __restrict__ W1, const float* __restrict__ b1,
                                   const float* __restrict__ W2, const float* __restrict__ b2,
                                   const float* __restrict__ W3, const float* __restrict__ b3,
                                   const float* __restrict__ W4, const float* __restrict__ b4,
                                   float* __restrict__ base) {
  long i = (long)blockIdx.x * 256 + threadIdx.x;            // 8*kNET total
  if (i >= 8 * kNET) return;
  int  e = (int)(i / kNET);
  long p = i - (long)e * kNET;
  float v;
  if      (p < OB1) v = W1[(long)e * 16384 + p];
  else if (p < OW2) v = b1[e * 256 + (p - OB1)];
  else if (p < OB2) v = W2[(long)e * 65536 + (p - OW2)];
  else if (p < OW3) v = b2[e * 256 + (p - OB2)];
  else if (p < OB3) v = W3[(long)e * 65536 + (p - OW3)];
  else if (p < OW4) v = b3[e * 256 + (p - OB3)];
  else if (p < OB4) v = W4[(long)e * 16384 + (p - OW4)];
  else              v = b4[e * 64 + (p - OB4)];
  base[i] = v;
}

__global__ void convert_y_kernel(const float* __restrict__ y,
                                 unsigned short* __restrict__ ybf) {
  long i = (long)blockIdx.x * 256 + threadIdx.x;
  ybf[i] = f2bfu(y[i]);
}

__global__ void gate_kernel(const float* __restrict__ ctx,
                            const float* __restrict__ G,
                            float* __restrict__ gate) {
  int b = threadIdx.x;
  if (b >= kENVS) return;
  float lg[kNEXP];
  float mx = -3.0e38f;
#pragma unroll
  for (int e = 0; e < kNEXP; ++e) {
    float s = 0.f;
    for (int c = 0; c < kCTX; ++c) s += ctx[b * kCTX + c] * G[e * kCTX + c];
    lg[e] = s;
    mx = fmaxf(mx, s);
  }
  float sum = 0.f;
#pragma unroll
  for (int e = 0; e < kNEXP; ++e) { lg[e] = __expf(lg[e] - mx); sum += lg[e]; }
  float inv = 1.f / sum;
#pragma unroll
  for (int e = 0; e < kNEXP; ++e) gate[b * kNEXP + e] = lg[e] * inv;
}

// ---------------------------------------------------------------- stage A:
// fW[(env*8+e)*NET + p] = bf16( base[e*NET+p] + dot(H[e,p,:], ctx[env,:]) )
// Block: stage 128 H rows (64KB, coalesced async b128) into padded LDS;
// each of 8 waves: 16 rows x 16 envs, K=128 -> 4 WMMAs. H read exactly once.
__global__ __launch_bounds__(256)
void fuse_weights_kernel(const float* __restrict__ H,
                         const float* __restrict__ ctx,
                         const float* __restrict__ base,
                         unsigned short* __restrict__ fW) {
  constexpr int LDH = kCTX + 4;                 // +4 dwords: bank-conflict-free
  __shared__ float lds_h[128 * LDH];

  const long g0b = (long)blockIdx.x * 128;
  // coalesced staging: 128 rows x 32 b128 copies
  for (int i = threadIdx.x; i < 128 * 32; i += 256) {
    int r = i >> 5, c = i & 31;
    copy16g(H + (size_t)(g0b + r) * kCTX + c * 4, lds_h + r * LDH + c * 4);
  }

  const int lane = threadIdx.x & 31;
  const int wave = threadIdx.x >> 5;
  const long g0 = g0b + wave * 16;
  const int n  = lane & 15;
  const int hi = lane >> 4;

  // B fragments: B[k][env] = ctx[env][k], 4 chunks of K=32 (tiny, from global)
  v16bf bfr[4];
#pragma unroll
  for (int c = 0; c < 4; ++c) {
    unsigned* bp = reinterpret_cast<unsigned*>(&bfr[c]);
    const float* crow = ctx + n * kCTX;
#pragma unroll
    for (int ii = 0; ii < 8; ++ii) {
      int k = ((ii * 2) & 7) + (((ii * 2) >> 3) << 4) + (hi << 3) + c * 32;
      bp[ii] = pack_bf2(crow[k], crow[k + 1]);
    }
  }

  wait_async_lds();
  __syncthreads();

  const float* row = lds_h + (wave * 16 + (lane & 15)) * LDH;
  v8f acc = {};
#pragma unroll
  for (int c = 0; c < 4; ++c) {
    v16bf a;
    unsigned* ap = reinterpret_cast<unsigned*>(&a);
#pragma unroll
    for (int ii = 0; ii < 8; ++ii) {
      int i = ii * 2;
      int k = (i & 7) + ((i >> 3) << 4) + (hi << 3) + c * 32;
      float2 v2 = *reinterpret_cast<const float2*>(row + k);
      ap[ii] = pack_bf2(v2.x, v2.y);
    }
    acc = __builtin_amdgcn_wmma_f32_16x16x32_bf16(false, a, false, bfr[c],
                                                  (short)0, acc, false, false);
  }

  // tiles never straddle experts (16 | kNET)
  const int  e  = (int)(g0 / kNET);
  const long p0 = g0 - (long)e * kNET;
#pragma unroll
  for (int j = 0; j < 8; ++j) {
    int   m = j + (hi << 3);
    float v = acc[j] + base[g0 + m];
    size_t idx = (size_t)(n * kNEXP + e) * kNET + p0 + m;
    fW[idx] = f2bfu(v);    // consecutive p -> vectorizes to global_store_b128
  }
}

// ---------------------------------------------------------------- stage B:
// Out[tok, nout] = act( X @ W^T + bias ).  Block = 128 tokens x 64 outs,
// 8 waves; each wave: one A frag -> 4 WMMAs (B-tile reuse). Padded LDS.
template <int KIN, bool ACT, bool XSHARED>
__global__ __launch_bounds__(256)
void mlp_layer_kernel(const unsigned short* __restrict__ X,
                      const unsigned short* __restrict__ fW,
                      long w_off, long b_off,
                      const float* __restrict__ beta,
                      unsigned short* __restrict__ Y,
                      int pair0, int nout) {
  constexpr int LDX = KIN + 8;                  // +8 bf16: bank-conflict-free
  constexpr int CPR = KIN / 8;                  // b128 copies per row
  __shared__ unsigned short lds_x[128 * LDX];
  __shared__ unsigned short lds_w[64 * LDX];

  const int pair_l = blockIdx.z;
  const int pair_g = pair0 + pair_l;
  const int ttile  = blockIdx.x;          // token tile (128 rows)
  const int otile  = blockIdx.y;          // out tile (64 cols)

  const unsigned short* Xp =
      X + (size_t)(XSHARED ? (pair_g >> 3) : pair_l) * kNTOK * KIN +
      (size_t)ttile * 128 * KIN;
  const unsigned short* Wp =
      fW + (size_t)pair_g * kNET + w_off + (size_t)otile * 64 * KIN;

  for (int i = threadIdx.x; i < 128 * CPR; i += 256) {
    int r = i / CPR, c = i % CPR;
    copy16g(Xp + r * KIN + c * 8, lds_x + r * LDX + c * 8);
  }
  for (int i = threadIdx.x; i < 64 * CPR; i += 256) {
    int r = i / CPR, c = i % CPR;
    copy16g(Wp + r * KIN + c * 8, lds_w + r * LDX + c * 8);
  }
  wait_async_lds();
  __syncthreads();

  const int lane = threadIdx.x & 31;
  const int wave = threadIdx.x >> 5;

  v8f acc[4] = {};
#pragma unroll
  for (int c = 0; c < KIN / 32; ++c) {
    v16bf a = load_a_frag(lds_x + wave * 16 * LDX, LDX, lane, c * 32);
#pragma unroll
    for (int t = 0; t < 4; ++t) {
      v16bf b = load_b_frag(lds_w + t * 16 * LDX, LDX, lane, c * 32);
      acc[t] = __builtin_amdgcn_wmma_f32_16x16x32_bf16(false, a, false, b,
                                                       (short)0, acc[t],
                                                       false, false);
    }
  }

  const int nn = lane & 15;
  const int hi = lane >> 4;
  const float bet = ACT ? beta[pair_g & 7] : 0.f;
  const size_t ybase = (size_t)pair_l * kNTOK * nout;
#pragma unroll
  for (int t = 0; t < 4; ++t) {
    const int col = otile * 64 + t * 16 + nn;
    const float bv = bfu2f(fW[(size_t)pair_g * kNET + b_off + col]);
#pragma unroll
    for (int j = 0; j < 8; ++j) {
      int   m   = j + (hi << 3);
      int   tok = ttile * 128 + wave * 16 + m;
      float v   = acc[t][j] + bv;
      if (ACT) v = v / (1.f + __expf(-bet * v));   // x * sigmoid(beta*x)
      Y[ybase + (size_t)tok * nout + col] = f2bfu(v);
    }
  }
}

// ---------------------------------------------------------------- final:
// out[b,tok,d] = sum_e gate[b,e] * ( h3[b,e,tok,:] @ W4^T + b4 )
// Block = 128 tokens x all 64 outs; loops experts in-block (no atomics).
__global__ __launch_bounds__(256)
void mlp_final_kernel(const unsigned short* __restrict__ X,   // chunk-local h3
                      const unsigned short* __restrict__ fW,
                      const float* __restrict__ gate,
                      float* __restrict__ out,
                      int env0) {
  constexpr int LDX = 256 + 8;
  __shared__ unsigned short lds_x[128 * LDX];
  __shared__ unsigned short lds_w[64 * LDX];

  const int env_l = blockIdx.z;
  const int env_g = env0 + env_l;
  const int ttile = blockIdx.x;
  const int lane  = threadIdx.x & 31;
  const int wave  = threadIdx.x >> 5;
  const int nn    = lane & 15;
  const int hi    = lane >> 4;

  v8f acc[4] = {};
#pragma unroll 1
  for (int e = 0; e < kNEXP; ++e) {
    const int pair_l = env_l * kNEXP + e;
    const int pair_g = env_g * kNEXP + e;
    const unsigned short* Xp =
        X + (size_t)pair_l * kNTOK * 256 + (size_t)ttile * 128 * 256;
    const unsigned short* Wp = fW + (size_t)pair_g * kNET + OW4;

    __syncthreads();
    for (int i = threadIdx.x; i < 128 * 32; i += 256) {
      int r = i >> 5, c = i & 31;
      copy16g(Xp + r * 256 + c * 8, lds_x + r * LDX + c * 8);
    }
    for (int i = threadIdx.x; i < 64 * 32; i += 256) {
      int r = i >> 5, c = i & 31;
      copy16g(Wp + r * 256 + c * 8, lds_w + r * LDX + c * 8);
    }
    wait_async_lds();
    __syncthreads();

    v8f ce[4] = {};
#pragma unroll
    for (int c = 0; c < 8; ++c) {
      v16bf a = load_a_frag(lds_x + wave * 16 * LDX, LDX, lane, c * 32);
#pragma unroll
      for (int t = 0; t < 4; ++t) {
        v16bf b = load_b_frag(lds_w + t * 16 * LDX, LDX, lane, c * 32);
        ce[t] = __builtin_amdgcn_wmma_f32_16x16x32_bf16(false, a, false, b,
                                                        (short)0, ce[t],
                                                        false, false);
      }
    }
    const float g = gate[pair_g];
#pragma unroll
    for (int t = 0; t < 4; ++t) {
      const float bv = bfu2f(fW[(size_t)pair_g * kNET + OB4 + t * 16 + nn]);
#pragma unroll
      for (int j = 0; j < 8; ++j) acc[t][j] += g * (ce[t][j] + bv);
    }
  }

#pragma unroll
  for (int t = 0; t < 4; ++t)
#pragma unroll
    for (int j = 0; j < 8; ++j) {
      int tok = ttile * 128 + wave * 16 + j + (hi << 3);
      out[((size_t)env_g * kNTOK + tok) * kDATA + t * 16 + nn] = acc[t][j];
    }
}

// ---------------------------------------------------------------- launch
extern "C" void kernel_launch(void* const* d_in, const int* in_sizes, int n_in,
                              void* d_out, int out_size, void* d_ws, size_t ws_size,
                              hipStream_t stream) {
  const float* y    = (const float*)d_in[1];
  const float* ctx  = (const float*)d_in[2];
  const float* W1   = (const float*)d_in[3];
  const float* b1   = (const float*)d_in[4];
  const float* W2   = (const float*)d_in[5];
  const float* b2   = (const float*)d_in[6];
  const float* W3   = (const float*)d_in[7];
  const float* b3   = (const float*)d_in[8];
  const float* W4   = (const float*)d_in[9];
  const float* b4   = (const float*)d_in[10];
  const float* H    = (const float*)d_in[11];
  const float* G    = (const float*)d_in[12];
  const float* beta = (const float*)d_in[13];
  float* out = (float*)d_out;

  char* ws = (char*)d_ws;
  float*          gate    = (float*)(ws + 0);                  //     512 B
  float*          basecat = (float*)(ws + 512);                // 5.27 MB
  unsigned short* ybf     = (unsigned short*)(ws + 5270016);   //  4.0 MB
  unsigned short* fW      = (unsigned short*)(ws + 9464320);   // 42.2 MB
  unsigned short* hA      = (unsigned short*)(ws + 51620352);  // 32 MB (32 pairs)
  unsigned short* hB      = (unsigned short*)(ws + 85174784);  // 32 MB

  // prep
  concat_base_kernel<<<(unsigned)((8 * kNET) / 256), 256, 0, stream>>>(
      W1, b1, W2, b2, W3, b3, W4, b4, basecat);
  convert_y_kernel<<<(kENVS * kNTOK * kDATA) / 256, 256, 0, stream>>>(y, ybf);
  gate_kernel<<<1, 64, 0, stream>>>(ctx, G, gate);

  // stage A: one pass over 674 MB of H, all 16 envs per row tile
  fuse_weights_kernel<<<(unsigned)((8 * kNET) / 128), 256, 0, stream>>>(
      H, ctx, basecat, fW);

  // stage B: env-groups of 4 (32 pairs) to bound workspace
  for (int g = 0; g < 4; ++g) {
    const int pair0 = g * 32;
    const int env0  = g * 4;
    mlp_layer_kernel<64, true, true>
        <<<dim3(16, 4, 32), 256, 0, stream>>>(ybf, fW, OW1, OB1, beta, hA,
                                              pair0, 256);
    mlp_layer_kernel<256, true, false>
        <<<dim3(16, 4, 32), 256, 0, stream>>>(hA, fW, OW2, OB2, beta, hB,
                                              pair0, 256);
    mlp_layer_kernel<256, true, false>
        <<<dim3(16, 4, 32), 256, 0, stream>>>(hB, fW, OW3, OB3, beta, hA,
                                              pair0, 256);
    mlp_final_kernel<<<dim3(16, 1, 4), 256, 0, stream>>>(hA, fW, gate, out,
                                                         env0);
  }
}
